// UBlock_90898687852949
// MI455X (gfx1250) — matrix operand order repaired and checked by reading, not compiled, not run on hardware
//
#include <hip/hip_runtime.h>

typedef __attribute__((ext_vector_type(16))) _Float16 v16h;
typedef __attribute__((ext_vector_type(8)))  float    v8f;
typedef __attribute__((ext_vector_type(4)))  int      v4i;
typedef _Float16 half_t;

#define EPSV 1e-4f
#define AS1 __attribute__((address_space(1)))
#define AS3 __attribute__((address_space(3)))

#if __has_builtin(__builtin_amdgcn_global_load_async_to_lds_b128) && \
    __has_builtin(__builtin_amdgcn_s_wait_asynccnt)
#define USE_ASYNC_LDS 1
#else
#define USE_ASYNC_LDS 0
#endif

// ------------------------------------------------------------------
// elementwise helpers
// ------------------------------------------------------------------
__global__ void k_zero(float* p, int n) {
  int i = blockIdx.x * 256 + threadIdx.x;
  if (i < n) p[i] = 0.f;
}

// convert f32 weights [KOFF][CIN][COUT] -> f16 transposed [KOFF][COUT][CIN]
__global__ void k_cvt_tr_f16(const float* __restrict__ w, half_t* __restrict__ o,
                             int CIN, int COUT, int n) {
  int i = blockIdx.x * 256 + threadIdx.x;   // dst index [k][c][kk]
  if (i < n) {
    int cc = CIN * COUT;
    int k = i / cc;
    int r = i - k * cc;
    int c = r / CIN;
    int kk = r - c * CIN;
    o[i] = (half_t)w[(size_t)k * cc + (size_t)kk * COUT + c];
  }
}

// partial column sums/sumsq with block reduction + global f32 atomics.
// requires blockDim==256, C in {32,64}
__global__ void k_colstats(const float* __restrict__ x, int n, int C, float* __restrict__ accum) {
  int tid = threadIdx.x;
  float s = 0.f, q = 0.f;
  for (int i = blockIdx.x * 256 + tid; i < n; i += gridDim.x * 256) {
    float v = x[i];
    s += v; q += v * v;
  }
  __shared__ float sh[512];
  sh[tid] = s; sh[256 + tid] = q;
  __syncthreads();
  for (int st = 128; st >= C; st >>= 1) {
    if (tid < st) { sh[tid] += sh[tid + st]; sh[256 + tid] += sh[256 + tid + st]; }
    __syncthreads();
  }
  if (tid < C) {
    atomicAdd(&accum[tid], sh[tid]);
    atomicAdd(&accum[C + tid], sh[256 + tid]);
  }
}

__global__ void k_bnparam(const float* __restrict__ accum, const float* __restrict__ g,
                          const float* __restrict__ b, float invN, int C, float* __restrict__ sb) {
  int c = threadIdx.x;
  if (c < C) {
    float mean = accum[c] * invN;
    float var  = accum[C + c] * invN - mean * mean;
    float s    = g[c] * rsqrtf(var + EPSV);
    sb[c]     = s;
    sb[C + c] = b[c] - mean * s;
  }
}

// BN(+ReLU) -> f16 (or plain f32->f16 when BN=0)
template <int C, int RELU, int BN>
__global__ void k_act_f16(const float* __restrict__ x, const float* __restrict__ sb,
                          int n, half_t* __restrict__ h) {
  int i = blockIdx.x * 256 + threadIdx.x;
  if (i < n) {
    int c = i & (C - 1);
    float v = x[i];
    if (BN)   v = v * sb[c] + sb[C + c];
    if (RELU) v = fmaxf(v, 0.f);
    h[i] = (half_t)v;
  }
}

__global__ void k_concat(const float* __restrict__ a, const float* __restrict__ b,
                         int N, float* __restrict__ o) {
  int i = blockIdx.x * 256 + threadIdx.x;
  if (i < N * 64) {
    int r = i >> 6, c = i & 63;
    o[i] = (c < 32) ? a[r * 32 + c] : b[r * 32 + (c - 32)];
  }
}

// ------------------------------------------------------------------
// WMMA gather-convolution.
//   out[m, :] (+= resid) = sum_k gather(h16, idx(m,k)) @ W[k]
// MODE 0: idx = nbr[m*KOFF+k]   (rulebook gconv, -1 = missing)
// MODE 1: idx = m               (dense row-wise matmul, KOFF==1)
// MODE 2: idx = (off[m]==k) ? nbr[m] : -1   (deconv: nbr=parent)
// Weights Wf16 pre-transposed: [KOFF][COUT][CIN] f16.
// 256 threads = 8 waves; each wave computes one 16-row tile x COUT.
// ------------------------------------------------------------------
template <int CIN, int COUT, int KOFF, int MODE>
__global__ void __launch_bounds__(256)
k_gconv(const half_t* __restrict__ h16, const int* __restrict__ nbr,
        const int* __restrict__ off, const half_t* __restrict__ Wf16,
        const float* __restrict__ resid, const half_t* __restrict__ zrow,
        float* __restrict__ out, int Nout)
{
  __shared__ __align__(16) half_t sW[CIN * COUT];       // [COUT][CIN]
  __shared__ __align__(16) half_t sA[8][16 * CIN];      // per-wave A tile

  const int tid  = threadIdx.x;
  const int wave = tid >> 5;
  const int lane = tid & 31;
  const int row  = lane & 15;        // M-row within tile (also B column within n-tile)
  const int hh   = lane >> 4;        // which half of the row this lane gathers
  const int mBase = (blockIdx.x * 8 + wave) * 16;
  const int m = mBase + row;

  v8f acc[COUT / 16] = {};

  for (int k = 0; k < KOFF; ++k) {
    __syncthreads();

    // ---- stage W[k] (already [COUT][CIN]) into LDS: contiguous b128 copy ----
#if USE_ASYNC_LDS
    {
      AS1 v4i* wsrc = (AS1 v4i*)(Wf16 + (size_t)k * (CIN * COUT));
      AS3 v4i* wdst = (AS3 v4i*)sW;
      for (int i = tid; i < (CIN * COUT) / 8; i += 256)
        __builtin_amdgcn_global_load_async_to_lds_b128(wsrc + i, wdst + i, 0, 0);
    }
#else
    {
      const uint4* wsrc = (const uint4*)(Wf16 + (size_t)k * (CIN * COUT));
      uint4*       wdst = (uint4*)sW;
      for (int i = tid; i < (CIN * COUT) / 8; i += 256) wdst[i] = wsrc[i];
    }
#endif

    // ---- gather one half neighbor row per lane into LDS A tile ----
    int j = -1;
    if (m < Nout) {
      if (MODE == 0)      j = nbr[(size_t)m * KOFF + k];
      else if (MODE == 1) j = m;
      else                j = (off[m] == k) ? nbr[m] : -1;
    }
#if USE_ASYNC_LDS
    {
      const half_t* srcg = (j >= 0) ? (h16 + (size_t)j * CIN + hh * (CIN / 2)) : zrow;
      AS1 v4i* gs = (AS1 v4i*)srcg;
      AS3 v4i* ld = (AS3 v4i*)&sA[wave][row * CIN + hh * (CIN / 2)];
#pragma unroll
      for (int t = 0; t < CIN / 16; ++t)
        __builtin_amdgcn_global_load_async_to_lds_b128(gs + t, ld + t, 0, 0);
    }
#else
    {
      uint4* dst = (uint4*)&sA[wave][row * CIN + hh * (CIN / 2)];
      if (j >= 0) {
        const uint4* src = (const uint4*)&h16[(size_t)j * CIN + hh * (CIN / 2)];
#pragma unroll
        for (int t = 0; t < CIN / 16; ++t) dst[t] = src[t];
      } else {
        uint4 zz{0u, 0u, 0u, 0u};
#pragma unroll
        for (int t = 0; t < CIN / 16; ++t) dst[t] = zz;
      }
    }
#endif
    // prefetch next offset's neighbor row (global_prefetch_b8)
    if (MODE == 0 && k + 1 < KOFF && m < Nout) {
      int jn = nbr[(size_t)m * KOFF + k + 1];
      if (jn >= 0) __builtin_prefetch(&h16[(size_t)jn * CIN], 0, 0);
    }
#if USE_ASYNC_LDS
    __builtin_amdgcn_s_wait_asynccnt(0);   // drain this wave's async LDS writes
#endif
    __syncthreads();

#pragma unroll
    for (int kc = 0; kc < CIN / 32; ++kc) {
      // A fragment: lane<16 -> K 0..7 & 16..23 ; lane>=16 -> K 8..15 & 24..31 (ISA layout)
      union { v16h h; unsigned u[8]; } A;
      const unsigned* arow = (const unsigned*)&sA[wave][row * CIN];
      const int kb = kc * 32 + (lane < 16 ? 0 : 8);
#pragma unroll
      for (int t = 0; t < 4; ++t) {
        A.u[t]     = arow[(kb >> 1) + t];
        A.u[4 + t] = arow[((kb + 16) >> 1) + t];
      }
#pragma unroll
      for (int nt = 0; nt < COUT / 16; ++nt) {
        // B fragment: lane = column; lanes 0-15 hold K 0..7 & 16..23, lanes 16-31 hold K 8..15 & 24..31
        union { v16h h; unsigned u[8]; } B;
        const int c = nt * 16 + row;
        const int kbase = kc * 32 + (lane < 16 ? 0 : 8);
        const unsigned* wrow = (const unsigned*)&sW[c * CIN + kbase];
#pragma unroll
        for (int t = 0; t < 4; ++t) { B.u[t] = wrow[t]; B.u[4 + t] = wrow[8 + t]; }
        acc[nt] = __builtin_amdgcn_wmma_f32_16x16x32_f16(
            false, A.h, false, B.h, (short)0, acc[nt], false, false);
      }
    }
  }

  // C/D layout: VGPR r -> M = r (lanes 0-15) / r+8 (lanes 16-31); N = lane&15
#pragma unroll
  for (int nt = 0; nt < COUT / 16; ++nt) {
#pragma unroll
    for (int r = 0; r < 8; ++r) {
      int mm = mBase + r + (lane < 16 ? 0 : 8);
      int c  = nt * 16 + row;
      if (mm < Nout) {
        size_t o = (size_t)mm * COUT + c;
        float v = acc[nt][r];
        out[o] = resid ? (v + resid[o]) : v;
      }
    }
  }
}

// ------------------------------------------------------------------
// host driver
// ------------------------------------------------------------------
extern "C" void kernel_launch(void* const* d_in, const int* in_sizes, int n_in,
                              void* d_out, int out_size, void* d_ws, size_t ws_size,
                              hipStream_t stream)
{
  (void)n_in; (void)out_size; (void)ws_size;
  const float* feats  = (const float*)d_in[0];
  const int*   nbr0   = (const int*)d_in[1];
  const int*   nbr1   = (const int*)d_in[2];
  const int*   up_nbr = (const int*)d_in[3];
  const int*   parent = (const int*)d_in[4];
  const int*   offp   = (const int*)d_in[5];
  const int N0 = in_sizes[4];
  const int N1 = in_sizes[2] / 27;

  char* ws = (char*)d_ws;
  size_t wo = 0;
  auto alloc = [&](size_t bytes) -> void* {
    void* p = ws + wo;
    wo = (wo + bytes + 255) & ~(size_t)255;
    return p;
  };

  float*  bufA   = (float*)alloc((size_t)N0 * 64 * sizeof(float));
  float*  bufB   = (float*)alloc((size_t)N0 * 64 * sizeof(float));
  float*  bufC   = (float*)alloc((size_t)N0 * 64 * sizeof(float));
  half_t* h16    = (half_t*)alloc((size_t)N0 * 64 * sizeof(half_t));
  float*  dA     = (float*)alloc((size_t)N1 * 64 * sizeof(float));
  float*  dB     = (float*)alloc((size_t)N1 * 64 * sizeof(float));
  half_t* dh16   = (half_t*)alloc((size_t)N1 * 64 * sizeof(half_t));
  float*  identb = (float*)alloc((size_t)N0 * 32 * sizeof(float));
  float*  upb    = (float*)alloc((size_t)N0 * 32 * sizeof(float));
  float*  accum  = (float*)alloc(512);
  float*  sb     = (float*)alloc(512);
  float*  zrowf  = (float*)alloc(256);
  const half_t* zrow = (const half_t*)zrowf;

  k_zero<<<1, 256, 0, stream>>>(zrowf, 64);   // 256B of zeros for missing neighbors

  auto cvt = [&](int idx, int K, int Ci, int Co) -> half_t* {
    int n = K * Ci * Co;
    half_t* dst = (half_t*)alloc((size_t)n * sizeof(half_t));
    k_cvt_tr_f16<<<(n + 255) / 256, 256, 0, stream>>>((const float*)d_in[idx], dst, Ci, Co, n);
    return dst;
  };
  half_t* Wb01 = cvt(8,  27, 32, 32);
  half_t* Wb02 = cvt(11, 27, 32, 32);
  half_t* Wb11 = cvt(14, 27, 32, 32);
  half_t* Wb12 = cvt(17, 27, 32, 32);
  half_t* Wdn  = cvt(20, 8,  32, 64);
  half_t* Wu01 = cvt(23, 27, 64, 64);
  half_t* Wu02 = cvt(26, 27, 64, 64);
  half_t* Wu11 = cvt(29, 27, 64, 64);
  half_t* Wu12 = cvt(32, 27, 64, 64);
  half_t* Wup  = cvt(35, 8,  64, 32);
  half_t* Wt01 = cvt(38, 27, 64, 32);
  half_t* Wt02 = cvt(41, 27, 32, 32);
  half_t* Pj   = cvt(42, 1,  64, 32);
  half_t* Wt11 = cvt(45, 27, 32, 32);
  half_t* Wt12 = cvt(48, 27, 32, 32);

  auto stats = [&](const float* x, int N, int C, int gi, int bi) {
    k_zero<<<1, 256, 0, stream>>>(accum, 2 * C);
    int n = N * C;
    int blocks = (n + 255) / 256; if (blocks > 512) blocks = 512;
    k_colstats<<<blocks, 256, 0, stream>>>(x, n, C, accum);
    k_bnparam<<<1, 128, 0, stream>>>(accum, (const float*)d_in[gi], (const float*)d_in[bi],
                                     1.0f / (float)N, C, sb);
  };
  auto act32 = [&](const float* x, half_t* h, int N) {
    int n = N * 32; k_act_f16<32, 1, 1><<<(n + 255) / 256, 256, 0, stream>>>(x, sb, n, h);
  };
  auto act64 = [&](const float* x, half_t* h, int N) {
    int n = N * 64; k_act_f16<64, 1, 1><<<(n + 255) / 256, 256, 0, stream>>>(x, sb, n, h);
  };

  const int g0 = (N0 + 127) / 128;
  const int g1 = (N1 + 127) / 128;

  // ---- blk0 (32 -> 32, nbr0) ----
  stats(feats, N0, 32, 6, 7);
  act32(feats, h16, N0);
  k_gconv<32, 32, 27, 0><<<g0, 256, 0, stream>>>(h16, nbr0, nullptr, Wb01, nullptr, zrow, bufB, N0);
  stats(bufB, N0, 32, 9, 10);
  act32(bufB, h16, N0);
  k_gconv<32, 32, 27, 0><<<g0, 256, 0, stream>>>(h16, nbr0, nullptr, Wb02, feats, zrow, bufA, N0);

  // ---- blk1 ----
  stats(bufA, N0, 32, 12, 13);
  act32(bufA, h16, N0);
  k_gconv<32, 32, 27, 0><<<g0, 256, 0, stream>>>(h16, nbr0, nullptr, Wb11, nullptr, zrow, bufB, N0);
  stats(bufB, N0, 32, 15, 16);
  act32(bufB, h16, N0);
  k_gconv<32, 32, 27, 0><<<g0, 256, 0, stream>>>(h16, nbr0, nullptr, Wb12, bufA, zrow, bufB, N0);
  // identity = bufB

  // ---- downsample conv (32 -> 64, up_nbr, N1 rows) ----
  stats(bufB, N0, 32, 18, 19);
  act32(bufB, h16, N0);
  k_gconv<32, 64, 8, 0><<<g1, 256, 0, stream>>>(h16, up_nbr, nullptr, Wdn, nullptr, zrow, dA, N1);

  // ---- ublk0 (64 -> 64, nbr1) ----
  stats(dA, N1, 64, 21, 22);
  act64(dA, dh16, N1);
  k_gconv<64, 64, 27, 0><<<g1, 256, 0, stream>>>(dh16, nbr1, nullptr, Wu01, nullptr, zrow, dB, N1);
  stats(dB, N1, 64, 24, 25);
  act64(dB, dh16, N1);
  k_gconv<64, 64, 27, 0><<<g1, 256, 0, stream>>>(dh16, nbr1, nullptr, Wu02, dA, zrow, dB, N1);

  // ---- ublk1 ----
  stats(dB, N1, 64, 27, 28);
  act64(dB, dh16, N1);
  k_gconv<64, 64, 27, 0><<<g1, 256, 0, stream>>>(dh16, nbr1, nullptr, Wu11, nullptr, zrow, dA, N1);
  stats(dA, N1, 64, 30, 31);
  act64(dA, dh16, N1);
  k_gconv<64, 64, 27, 0><<<g1, 256, 0, stream>>>(dh16, nbr1, nullptr, Wu12, dB, zrow, dA, N1);

  // ---- deconv (64 -> 32, scatter via parent/off) ----
  stats(dA, N1, 64, 33, 34);
  act64(dA, dh16, N1);
  k_gconv<64, 32, 8, 2><<<g0, 256, 0, stream>>>(dh16, parent, offp, Wup, nullptr, zrow, upb, N0);

  // ---- concat [identity(bufB,32) | up(upb,32)] -> bufC (N0 x 64) ----
  k_concat<<<(N0 * 64 + 255) / 256, 256, 0, stream>>>(bufB, upb, N0, bufC);

  // ---- tail0 (64 -> 32, projection) ----
  { int n = N0 * 64; k_act_f16<64, 0, 0><<<(n + 255) / 256, 256, 0, stream>>>(bufC, sb, n, h16); }
  k_gconv<64, 32, 1, 1><<<g0, 256, 0, stream>>>(h16, nullptr, nullptr, Pj, nullptr, zrow, identb, N0);
  stats(bufC, N0, 64, 36, 37);
  act64(bufC, h16, N0);
  k_gconv<64, 32, 27, 0><<<g0, 256, 0, stream>>>(h16, nbr0, nullptr, Wt01, nullptr, zrow, bufB, N0);
  stats(bufB, N0, 32, 39, 40);
  act32(bufB, h16, N0);
  k_gconv<32, 32, 27, 0><<<g0, 256, 0, stream>>>(h16, nbr0, nullptr, Wt02, identb, zrow, bufB, N0);

  // ---- tail1 (32 -> 32) ----
  stats(bufB, N0, 32, 43, 44);
  act32(bufB, h16, N0);
  k_gconv<32, 32, 27, 0><<<g0, 256, 0, stream>>>(h16, nbr0, nullptr, Wt11, nullptr, zrow, bufA, N0);
  stats(bufA, N0, 32, 46, 47);
  act32(bufA, h16, N0);
  k_gconv<32, 32, 27, 0><<<g0, 256, 0, stream>>>(h16, nbr0, nullptr, Wt12, bufB, zrow, (float*)d_out, N0);
}
